// MultiHeadAttention_2173253451924
// MI455X (gfx1250) — matrix operand order repaired and checked by reading, not compiled, not run on hardware
//
#include <hip/hip_runtime.h>
#include <hip/hip_bf16.h>

#define BATCH  2
#define SEQ    2048
#define DMODEL 1024
#define NHEAD  16
#define DHEAD  64

typedef __attribute__((ext_vector_type(16))) __bf16 v16bf;
typedef __attribute__((ext_vector_type(8)))  __bf16 v8bf;
typedef __attribute__((ext_vector_type(8)))  float  v8f;
typedef __attribute__((ext_vector_type(4)))  int    v4i;

// ---------------------------------------------------------------------------
// WMMA helpers (CDNA5 gfx1250, wave32). D = A(16x32 bf16) * B(32x16 bf16) + C(f32)
// ---------------------------------------------------------------------------
__device__ __forceinline__ v8f wmma_bf16(v16bf a, v16bf b, v8f c) {
    return __builtin_amdgcn_wmma_f32_16x16x32_bf16(
        /*neg_a=*/false, a, /*neg_b=*/false, b,
        /*c_mod=*/(short)0, c, /*reuse_a=*/false, /*reuse_b=*/false);
}

// A matrix 16x32 bf16 from row-major [rows x ld]:
// lane<16: M=lane, halves 0..15 = K k0+0..15
// lane>=16: M=lane-16, halves 0..15 = K k0+16..31
__device__ __forceinline__ v16bf load_fragA(const __bf16* base, int ld, int m0, int k0) {
    int lane = threadIdx.x & 31;
    int m    = m0 + (lane & 15);
    int koff = (lane >> 4) << 4;                 // 0 or 16
    const __bf16* p = base + (size_t)m * ld + k0 + koff;
    v8bf lo = *reinterpret_cast<const v8bf*>(p);
    v8bf hi = *reinterpret_cast<const v8bf*>(p + 8);
    v16bf f;
#pragma unroll
    for (int i = 0; i < 8; ++i) { f[i] = lo[i]; f[8 + i] = hi[i]; }
    return f;
}

// B matrix 32x16 bf16 where B[k][n] = src[n*ld + k]  ("NT": contiguous in k)
// lane n = lane&15, lanes 0-15 hold K k0+0..15, lanes 16-31 hold K k0+16..31
__device__ __forceinline__ v16bf load_fragB_nt(const __bf16* base, int ld, int n0, int k0) {
    int lane = threadIdx.x & 31;
    int n    = n0 + (lane & 15);
    int koff = (lane >> 4) << 4;                 // 0 or 16
    const __bf16* p = base + (size_t)n * ld + k0 + koff;
    return *reinterpret_cast<const v16bf*>(p);
}

// B matrix 32x16 bf16 where B[k][n] = src[k*ld + n]  ("NN": strided in k)
__device__ __forceinline__ v16bf load_fragB_nn(const __bf16* base, int ld, int k0, int n0) {
    int lane = threadIdx.x & 31;
    int n    = n0 + (lane & 15);
    int k    = k0 + ((lane >> 4) << 4);
    const __bf16* p = base + (size_t)k * ld + n;
    v16bf f;
#pragma unroll
    for (int i = 0; i < 16; ++i) f[i] = p[(size_t)i * ld];
    return f;
}

// ---------------------------------------------------------------------------
// Async global -> LDS staging (CDNA5). Tracked by ASYNCcnt, decoupled from
// VGPRs. Falls back to a synchronous 16-byte copy if the builtin is absent.
// Builtin signature (from hipcc diagnostic): param0 = v4i in addrspace(1).
// ---------------------------------------------------------------------------
#if __has_builtin(__builtin_amdgcn_global_load_async_to_lds_b128)
#define HAVE_ASYNC_LDS 1
typedef __attribute__((address_space(1))) v4i* g_v4i_ptr;
typedef __attribute__((address_space(3))) v4i* l_v4i_ptr;
#else
#define HAVE_ASYNC_LDS 0
#endif

__device__ __forceinline__ void copy16_g2l(__bf16* lds, const __bf16* glob) {
#if HAVE_ASYNC_LDS
    __builtin_amdgcn_global_load_async_to_lds_b128(
        (g_v4i_ptr)(void*)const_cast<__bf16*>(glob),
        (l_v4i_ptr)(void*)lds,
        /*offset=*/0, /*cpol=*/0);
#else
    *reinterpret_cast<uint4*>(lds) = *reinterpret_cast<const uint4*>(glob);
#endif
}

__device__ __forceinline__ void wait_async_copies() {
#if HAVE_ASYNC_LDS && __has_builtin(__builtin_amdgcn_s_wait_asynccnt)
    __builtin_amdgcn_s_wait_asynccnt(0);
#endif
}

// A fragment of softmax probabilities, computed on the fly from f32 logits:
// p = exp(qk - rowmax), cast to bf16. qrow points at the 16-row tile base.
__device__ __forceinline__ v16bf make_pfrag(const float* qrow, int k0, float rm_lane) {
    int lane = threadIdx.x & 31;
    int m    = lane & 15;
    int koff = (lane >> 4) << 4;
    const float* p = qrow + (size_t)m * SEQ + k0 + koff;
    v16bf f;
#pragma unroll
    for (int i = 0; i < 8; ++i) {
        f[i]     = (__bf16)__expf(p[i]      - rm_lane);
        f[8 + i] = (__bf16)__expf(p[8 + i] - rm_lane);
    }
    return f;
}

// ---------------------------------------------------------------------------
// Kernel 1: f32 -> bf16 conversion
// ---------------------------------------------------------------------------
__global__ void cvt_f32_bf16(const float* __restrict__ src, __bf16* __restrict__ dst, int n) {
    int i = blockIdx.x * blockDim.x + threadIdx.x;
    if (i < n) dst[i] = (__bf16)src[i];
}

// ---------------------------------------------------------------------------
// Kernel 2: NT GEMM  out[m][n] = (sum_k A[m][k]*W[n][k] + bias[n]) * scale
// 256 threads = 8 waves (2 in M x 4 in N), block tile 32x256, BK=32.
// Double-buffered async LDS staging: K+1 tile streams into LDS (ASYNCcnt)
// while WMMAs consume the current tile from ds_load_b128.
// ---------------------------------------------------------------------------
#define BM 32
#define BN 256
#define BK 32

template <bool BF16OUT>
__global__ __launch_bounds__(256) void gemm_nt(const __bf16* __restrict__ A,
                                               const __bf16* __restrict__ W,
                                               const float*  __restrict__ bias,
                                               void* __restrict__ outv,
                                               int M, int N, int K, float scale) {
    __shared__ __bf16 sA[2][BM * BK];   // 2 x 2 KB
    __shared__ __bf16 sB[2][BN * BK];   // 2 x 16 KB

    int t    = threadIdx.x;
    int wave = t >> 5;
    int lane = t & 31;
    int mblk = blockIdx.y * BM;
    int nblk = blockIdx.x * BN;
    int m0   = (wave & 1) * 16;          // wave tile origin inside block tile
    int n0   = (wave >> 1) * 64;

    auto stage = [&](int buf, int k0) {
        // B tile: BN x BK bf16 = 1024 x 16B chunks, 4 per thread
#pragma unroll
        for (int i = 0; i < 4; ++i) {
            int c   = t * 4 + i;
            int row = c >> 2;            // 0..255
            int col = (c & 3) * 8;       // 0,8,16,24 (bf16 elements)
            copy16_g2l(&sB[buf][row * BK + col], W + (size_t)(nblk + row) * K + k0 + col);
        }
        // A tile: BM x BK bf16 = 128 x 16B chunks, threads 0..127
        if (t < 128) {
            int row = t >> 2;            // 0..31
            int col = (t & 3) * 8;
            copy16_g2l(&sA[buf][row * BK + col], A + (size_t)(mblk + row) * K + k0 + col);
        }
    };

    v8f acc[4] = {};
    stage(0, 0);
    const int NK = K / BK;
    for (int ks = 0; ks < NK; ++ks) {
        int cur = ks & 1;
        wait_async_copies();
        __syncthreads();                          // tile[cur] ready; prior reads done
        if (ks + 1 < NK) stage(cur ^ 1, (ks + 1) * BK);
        if (ks + 2 < NK)                          // pull K+2 tile toward L2
            __builtin_prefetch((const void*)(W + (size_t)(nblk + t) * K + (ks + 2) * BK), 0, 0);

        v16bf a = load_fragA(sA[cur], BK, m0, 0);
#pragma unroll
        for (int j = 0; j < 4; ++j)
            acc[j] = wmma_bf16(a, load_fragB_nt(sB[cur], BK, n0 + 16 * j, 0), acc[j]);
        __syncthreads();                          // done reading tile[cur]
    }

    int nlane = lane & 15;
    int mhi   = (lane >> 4) * 8;
#pragma unroll
    for (int j = 0; j < 4; ++j) {
        int   n  = nblk + n0 + 16 * j + nlane;
        float bb = bias ? bias[n] : 0.0f;
#pragma unroll
        for (int r = 0; r < 8; ++r) {
            int   m = mblk + m0 + mhi + r;
            float v = (acc[j][r] + bb) * scale;
            if constexpr (BF16OUT) ((__bf16*)outv)[(size_t)m * N + n] = (__bf16)v;
            else                   ((float*)outv)[(size_t)m * N + n]  = v;
        }
    }
}

// ---------------------------------------------------------------------------
// Kernel 3: qk[b,h,m,n] = sum_d qs[m][d]*ks[n][d] + mask[m][n]   (scales folded)
// K = DHEAD = 64 -> exactly 2 K-steps: hoist all 10 fragments, then issue the
// 8 independent WMMAs back-to-back (no WMMA->WMMA accumulate hazard NOPs).
// ---------------------------------------------------------------------------
__global__ __launch_bounds__(256) void qk_kernel(const __bf16* __restrict__ qb,
                                                 const __bf16* __restrict__ kb,
                                                 const float*  __restrict__ mask,
                                                 float* __restrict__ qk) {
    int bh = blockIdx.z;
    int b  = bh >> 4;
    int h  = bh & 15;
    const __bf16* qptr = qb + ((size_t)b * SEQ) * DMODEL + h * DHEAD;
    const __bf16* kptr = kb + ((size_t)b * SEQ) * DMODEL + h * DHEAD;
    float* o = qk + (size_t)bh * SEQ * SEQ;

    int wave = threadIdx.x >> 5;
    int lane = threadIdx.x & 31;
    int m0   = blockIdx.y * 32  + (wave & 1) * 16;
    int n0   = blockIdx.x * 256 + (wave >> 1) * 64;

    // hoist all fragment loads
    v16bf a0 = load_fragA(qptr, DMODEL, m0, 0);
    v16bf a1 = load_fragA(qptr, DMODEL, m0, 32);
    v16bf b0[4], b1[4];
#pragma unroll
    for (int j = 0; j < 4; ++j) {
        b0[j] = load_fragB_nt(kptr, DMODEL, n0 + 16 * j, 0);
        b1[j] = load_fragB_nt(kptr, DMODEL, n0 + 16 * j, 32);
    }
    v8f acc[4] = {};
#pragma unroll
    for (int j = 0; j < 4; ++j) acc[j] = wmma_bf16(a0, b0[j], acc[j]);
#pragma unroll
    for (int j = 0; j < 4; ++j) acc[j] = wmma_bf16(a1, b1[j], acc[j]);

    int nlane = lane & 15;
    int mhi   = (lane >> 4) * 8;
#pragma unroll
    for (int j = 0; j < 4; ++j) {
        int n = n0 + 16 * j + nlane;
#pragma unroll
        for (int r = 0; r < 8; ++r) {
            int m = m0 + mhi + r;
            o[(size_t)m * SEQ + n] = acc[j][r] + mask[(size_t)m * SEQ + n];
        }
    }
}

// ---------------------------------------------------------------------------
// Kernel 4: fused softmax + P·V for a 16-row q tile of one (b,h).
// 128 threads = 4 waves; each wave covers SEQ/4 of K for all 4 dh sub-tiles,
// partials reduced through LDS, scaled by 1/rowsum, emitted bf16 (b,s,h*dh).
// ---------------------------------------------------------------------------
__global__ __launch_bounds__(128) void softmax_wv_kernel(const float*  __restrict__ qk,
                                                         const __bf16* __restrict__ vb,
                                                         __bf16* __restrict__ wvb) {
    __shared__ float red[128];
    __shared__ float rowmax[16];
    __shared__ float rowsum[16];
    __shared__ float partial[4][4][256];   // [wave][ntile][lane*8+i]

    int bh = blockIdx.y;
    int b  = bh >> 4;
    int h  = bh & 15;
    int q0 = blockIdx.x * 16;
    const float* qrow = qk + ((size_t)bh * SEQ + q0) * SEQ;

    int t = threadIdx.x;
    int r = t >> 3;       // row 0..15
    int c = t & 7;        // 8 threads per row

    // pass 1: row max (mask already folded into qk)
    float mx = -3.4e38f;
    for (int j = c; j < SEQ; j += 8) mx = fmaxf(mx, qrow[(size_t)r * SEQ + j]);
    red[t] = mx;
    __syncthreads();
    if (c == 0) {
        float v = red[r * 8];
        for (int i = 1; i < 8; ++i) v = fmaxf(v, red[r * 8 + i]);
        rowmax[r] = v;
    }
    __syncthreads();
    // pass 2: row sum of exp
    float rm = rowmax[r];
    float sm = 0.0f;
    for (int j = c; j < SEQ; j += 8) sm += __expf(qrow[(size_t)r * SEQ + j] - rm);
    red[t] = sm;
    __syncthreads();
    if (c == 0) {
        float v = 0.0f;
        for (int i = 0; i < 8; ++i) v += red[r * 8 + i];
        rowsum[r] = v;
    }
    __syncthreads();

    // P·V with on-the-fly exp recompute (A frag), K split across 4 waves
    int wave = t >> 5;
    int lane = t & 31;
    float rm_lane = rowmax[lane & 15];
    const __bf16* vptr = vb + ((size_t)b * SEQ) * DMODEL + h * DHEAD;
    v8f acc[4] = {};
    for (int k0 = wave * (SEQ / 4); k0 < (wave + 1) * (SEQ / 4); k0 += 32) {
        v16bf a = make_pfrag(qrow, k0, rm_lane);
#pragma unroll
        for (int j = 0; j < 4; ++j)
            acc[j] = wmma_bf16(a, load_fragB_nn(vptr, DMODEL, k0, j * 16), acc[j]);
    }
#pragma unroll
    for (int j = 0; j < 4; ++j)
#pragma unroll
        for (int i = 0; i < 8; ++i) partial[wave][j][lane * 8 + i] = acc[j][i];
    __syncthreads();

    // wave w finalizes dh sub-tile w
    int j     = wave;
    int nlane = lane & 15;
    int mhi   = (lane >> 4) * 8;
#pragma unroll
    for (int i = 0; i < 8; ++i) {
        float v = partial[0][j][lane * 8 + i] + partial[1][j][lane * 8 + i]
                + partial[2][j][lane * 8 + i] + partial[3][j][lane * 8 + i];
        int m = mhi + i;
        v = v / rowsum[m];
        wvb[((size_t)(b * SEQ + q0 + m)) * DMODEL + h * DHEAD + j * 16 + nlane] = (__bf16)v;
    }
}

// ---------------------------------------------------------------------------
// Host launcher
// ---------------------------------------------------------------------------
extern "C" void kernel_launch(void* const* d_in, const int* in_sizes, int n_in,
                              void* d_out, int out_size, void* d_ws, size_t ws_size,
                              hipStream_t stream) {
    (void)in_sizes; (void)n_in; (void)out_size; (void)ws_size;
    const float* x    = (const float*)d_in[0];
    const float* mask = (const float*)d_in[1];
    const float* Wq   = (const float*)d_in[2];
    const float* bq   = (const float*)d_in[3];
    const float* Wk   = (const float*)d_in[4];
    const float* Wv   = (const float*)d_in[5];
    const float* bv   = (const float*)d_in[6];
    const float* Wo   = (const float*)d_in[7];
    const float* bo   = (const float*)d_in[8];

    float* out = (float*)d_out;                                   // [B*S*D]
    float* qko = out + (size_t)BATCH * SEQ * DMODEL;              // [B*H*S*S]

    const size_t XE = (size_t)BATCH * SEQ * DMODEL;   // 4,194,304
    const size_t WE = (size_t)DMODEL * DMODEL;        // 1,048,576

    char*  ws  = (char*)d_ws;
    size_t off = 0;
    auto alloc = [&](size_t bytes) {
        char* p = ws + off;
        off += (bytes + 255) & ~(size_t)255;
        return p;
    };
    __bf16* xb   = (__bf16*)alloc(XE * 2);   // also reused for wv after attn
    __bf16* wqb  = (__bf16*)alloc(WE * 2);
    __bf16* wkb  = (__bf16*)alloc(WE * 2);
    __bf16* wvwb = (__bf16*)alloc(WE * 2);
    __bf16* wob  = (__bf16*)alloc(WE * 2);
    __bf16* qb   = (__bf16*)alloc(XE * 2);
    __bf16* kb   = (__bf16*)alloc(XE * 2);
    __bf16* vbuf = (__bf16*)alloc(XE * 2);

    const float scale = 0.35355339059327373f;   // 64^(-1/4)

    cvt_f32_bf16<<<(int)((XE + 255) / 256), 256, 0, stream>>>(x,  xb,   (int)XE);
    cvt_f32_bf16<<<(int)((WE + 255) / 256), 256, 0, stream>>>(Wq, wqb,  (int)WE);
    cvt_f32_bf16<<<(int)((WE + 255) / 256), 256, 0, stream>>>(Wk, wkb,  (int)WE);
    cvt_f32_bf16<<<(int)((WE + 255) / 256), 256, 0, stream>>>(Wv, wvwb, (int)WE);
    cvt_f32_bf16<<<(int)((WE + 255) / 256), 256, 0, stream>>>(Wo, wob,  (int)WE);

    dim3 gproj(DMODEL / BN, (BATCH * SEQ) / BM);                  // (4, 128)
    gemm_nt<true><<<gproj, 256, 0, stream>>>(xb, wqb,  bq,      qb,   BATCH * SEQ, DMODEL, DMODEL, scale);
    gemm_nt<true><<<gproj, 256, 0, stream>>>(xb, wkb,  nullptr, kb,   BATCH * SEQ, DMODEL, DMODEL, scale);
    gemm_nt<true><<<gproj, 256, 0, stream>>>(xb, wvwb, bv,      vbuf, BATCH * SEQ, DMODEL, DMODEL, 1.0f);

    qk_kernel<<<dim3(SEQ / 256, SEQ / 32, BATCH * NHEAD), 256, 0, stream>>>(qb, kb, mask, qko);

    // xb reused as the bf16 attention output (x no longer needed)
    softmax_wv_kernel<<<dim3(SEQ / 16, BATCH * NHEAD), 128, 0, stream>>>(qko, vbuf, xb);

    gemm_nt<false><<<gproj, 256, 0, stream>>>(xb, wob, bo, out, BATCH * SEQ, DMODEL, DMODEL, 1.0f);
}
